// BRN_72842645340791
// MI455X (gfx1250) — matrix operand order repaired and checked by reading, not compiled, not run on hardware
//
#include <hip/hip_runtime.h>
#include <stdint.h>
#include <math.h>

typedef __attribute__((ext_vector_type(16))) __bf16 v16bf;
typedef __attribute__((ext_vector_type(8)))  __bf16 v8bf;
typedef __attribute__((ext_vector_type(8)))  float  v8f;

#define NB   8      // batch
#define TT   4096   // time steps
#define DIM  1024
#define BDM  128
#define EPSV 1e-5f

// ---------------- WMMA helpers (CDNA5 v_wmma_f32_16x16x32_bf16) ----------------

__device__ __forceinline__ v8f wmma_bf16(v16bf a, v16bf b, v8f c) {
  return __builtin_amdgcn_wmma_f32_16x16x32_bf16(false, a, false, b, (short)0, c, false, false);
}

__device__ __forceinline__ v16bf cat8(v8bf lo, v8bf hi) {
  v16bf r;
#pragma unroll
  for (int i = 0; i < 8; ++i) { r[i] = lo[i]; r[i + 8] = hi[i]; }
  return r;
}

// A-matrix 16x32 bf16 fragment from row-major [16][stride] bf16 buffer.
// ISA layout: lanes 0-15 -> M=lane, K = 0..7 / 16..23; lanes 16-31 -> K = 8..15 / 24..31.
__device__ __forceinline__ v16bf load_a(const __bf16* A, int stride, int ks, int lane) {
  int m  = lane & 15;
  int kb = (lane < 16) ? 0 : 8;
  const __bf16* p = A + m * stride + ks * 32 + kb;
  v8bf lo = *(const v8bf*)p;
  v8bf hi = *(const v8bf*)(p + 16);
  return cat8(lo, hi);
}

// B-matrix 32x16 bf16 fragment so D = A*B computes inp @ W^T for row-major W[N][K]:
// lane n = lane&15 holds W[ntile*16+n][ks*32 + kb + 0..7 / 16..23], kb = (lane<16)?0:8.
__device__ __forceinline__ v16bf load_b(const __bf16* W, int stride, int ntile, int ks, int lane) {
  int n  = lane & 15;
  int kb = (lane < 16) ? 0 : 8;
  const __bf16* p = W + (size_t)(ntile * 16 + n) * stride + ks * 32 + kb;
  v8bf lo = *(const v8bf*)p;
  v8bf hi = *(const v8bf*)(p + 16);
  return cat8(lo, hi);
}

// Async global -> LDS copy (CDNA5 GLOBAL_LOAD_ASYNC_TO_LDS_B128, ASYNCcnt-tracked).
__device__ __forceinline__ void async_g2l_b128(uint32_t lds_off, const void* gptr) {
  uint64_t ga = (uint64_t)(uintptr_t)gptr;
  asm volatile("global_load_async_to_lds_b128 %0, %1, off"
               :: "v"(lds_off), "v"(ga) : "memory");
}
__device__ __forceinline__ void wait_asynccnt0() {
  asm volatile("s_wait_asynccnt 0" ::: "memory");
}

// ---------------- Kernel 0: Wp f32 -> bf16 (one-shot, 256 KB -> L2 resident) ----------------

__global__ __launch_bounds__(256) void wprep_kernel(const float* __restrict__ Wp,
                                                    __bf16* __restrict__ wpb) {
  int e = blockIdx.x * 256 + threadIdx.x;   // 128*1024 elements
  wpb[e] = (__bf16)Wp[e];
}

// ---------------- Kernel 1: h = x @ Wp^T  (HBM-bound, async-LDS staged, bf16 WMMA) ----------
// x: [8][4096][1024] f32, wpb: [128][1024] bf16, h: [4096][8][128] bf16

__global__ __launch_bounds__(256) void proj_kernel(const float* __restrict__ x,
                                                   const __bf16* __restrict__ wpb,
                                                   __bf16* __restrict__ h) {
  extern __shared__ char psm[];
  float*  sXf = (float*)psm;                  // [16][1024] f32  64 KB (async landing pad)
  __bf16* sXb = (__bf16*)(sXf + 16 * DIM);    // [16][1024] bf16 32 KB (WMMA A operand)

  const int tid  = threadIdx.x;
  const int lane = tid & 31;
  const int w    = tid >> 5;                  // 8 waves -> 8 N-tiles of 16
  const int t0   = blockIdx.x * 16;
  const int b    = blockIdx.y;

  // ---- async-stage the 64 KB x tile straight into LDS (no VGPR round trip) ----
  const char* xbase = (const char*)(x + ((size_t)b * TT + t0) * DIM);
  const uint32_t lbase = (uint32_t)(uintptr_t)sXf;
#pragma unroll
  for (int i = 0; i < 16; ++i) {              // 16 iters x 256 lanes x 16 B = 64 KB
    uint32_t off = i * 4096u + tid * 16u;
    async_g2l_b128(lbase + off, xbase + off);
  }
  wait_asynccnt0();
  __syncthreads();

  // convert once to bf16 (A fragments then load clean b128s)
  for (int e = tid; e < 16 * DIM; e += 256) sXb[e] = (__bf16)sXf[e];
  __syncthreads();

  // ---- software-pipelined WMMA K-loop: fetch ks+1 while multiplying ks ----
  v8f acc0 = {0.f, 0.f, 0.f, 0.f, 0.f, 0.f, 0.f, 0.f};
  v8f acc1 = {0.f, 0.f, 0.f, 0.f, 0.f, 0.f, 0.f, 0.f};
  v16bf a0 = load_a(sXb, DIM, 0, lane);
  v16bf b0 = load_b(wpb, DIM, w, 0, lane);
#pragma unroll 4
  for (int ks = 0; ks < DIM / 32 - 1; ++ks) {
    v16bf a1 = load_a(sXb, DIM, ks + 1, lane);
    v16bf b1 = load_b(wpb, DIM, w, ks + 1, lane);
    if (ks + 8 < DIM / 32)   // prefetch bf16 weight stream ahead (global_prefetch_b8)
      __builtin_prefetch(wpb + (size_t)(w * 16 + (lane & 15)) * DIM + (ks + 8) * 32, 0, 1);
    if (ks & 1) acc1 = wmma_bf16(a0, b0, acc1);
    else        acc0 = wmma_bf16(a0, b0, acc0);
    a0 = a1; b0 = b1;
  }
  acc1 = wmma_bf16(a0, b0, acc1);
  v8f acc = acc0 + acc1;

  const int n   = lane & 15;
  const int mo  = (lane < 16) ? 0 : 8;
  const int col = w * 16 + n;
#pragma unroll
  for (int r = 0; r < 8; ++r) {
    int m = r + mo;           // time row within tile
    h[((size_t)(t0 + m) * NB + b) * BDM + col] = (__bf16)acc[r];
  }
}

// ---------------- Kernel 2: the scan (single WGP, weights LDS-resident) ----------------
// One block, 256 threads (8 wave32). Dynamic LDS ~187 KB (CDNA5 WGP: 320 KB).

__global__ __launch_bounds__(256) void scan_kernel(const __bf16* __restrict__ h,
                                                   const float* __restrict__ Wg1,
                                                   const float* __restrict__ bg1,
                                                   const float* __restrict__ Wg2,
                                                   const float* __restrict__ bg2,
                                                   const float* __restrict__ Wu1,
                                                   const float* __restrict__ bu1,
                                                   const float* __restrict__ Wu2,
                                                   const float* __restrict__ bu2,
                                                   const float* __restrict__ gamma,
                                                   const float* __restrict__ beta,
                                                   float* __restrict__ out) {
  extern __shared__ char smem[];
  __bf16* sWg1 = (__bf16*)smem;            // [128][256]  64 KB
  __bf16* sWu1 = sWg1 + 128 * 256;         // [128][256]  64 KB
  __bf16* sWu2 = sWu1 + 128 * 256;         // [128][128]  32 KB
  __bf16* sInp = sWu2 + 128 * 128;         // [16][256]   8 KB   (belief | ht), bf16
  __bf16* sHg  = sInp + 16 * 256;          // [16][128]   4 KB
  __bf16* sHu  = sHg + 16 * 128;           // [16][128]   4 KB
  float*  sBel = (float*)(sHu + 16 * 128); // [16][128]   8 KB   f32 belief
  float*  sbg1 = sBel + 16 * 128;          // 128
  float*  sbu1 = sbg1 + 128;
  float*  sbu2 = sbu1 + 128;
  float*  swg2 = sbu2 + 128;
  float*  sgam = swg2 + 128;
  float*  sbet = sgam + 128;
  float*  sGate = sbet + 128;              // [16] gates + [16]=bg2 scalar

  const int tid  = threadIdx.x;
  const int lane = tid & 31;
  const int w    = tid >> 5;

  // ---- one-time staging: f32 weights -> bf16 LDS ----
  for (int e = tid; e < 128 * 256; e += 256) sWg1[e] = (__bf16)Wg1[e];
  for (int e = tid; e < 128 * 256; e += 256) sWu1[e] = (__bf16)Wu1[e];
  for (int e = tid; e < 128 * 128; e += 256) sWu2[e] = (__bf16)Wu2[e];
  if (tid < 128) {
    sbg1[tid] = bg1[tid];  sbu1[tid] = bu1[tid];  sbu2[tid] = bu2[tid];
    swg2[tid] = Wg2[tid];  sgam[tid] = gamma[tid]; sbet[tid] = beta[tid];
  }
  if (tid == 0) sGate[16] = bg2[0];
  for (int e = tid; e < 16 * 128; e += 256) sBel[e] = 0.0f;
  for (int e = tid; e < 16 * 256; e += 256) sInp[e] = (__bf16)0.0f;  // pad rows' h half stays 0
  // preload h_0 into inp[:, 128:256]
  if (tid < 128) {
    int e = tid * 8; int m = e >> 7, c = e & 127;
    *(v8bf*)(sInp + m * 256 + 128 + c) = *(const v8bf*)(h + (size_t)m * BDM + c);
  }
  __syncthreads();

  const float bg2v = sGate[16];
  const int   n    = lane & 15;
  const int   col  = w * 16 + n;
  const int   mo   = (lane < 16) ? 0 : 8;
  const int   mln  = w * 2 + ((lane < 16) ? 0 : 1);  // LayerNorm/gate row for this lane
  const int   c0   = (lane & 15) * 8;

  for (int t = 0; t < TT; ++t) {
    // ---- phase 1: hidden_g = relu(inp@Wg1^T+bg1), hidden_u = relu(inp@Wu1^T+bu1)
    // dual accumulators: WMMA dependency depth 4 instead of 8
    v8f accG0 = {0.f, 0.f, 0.f, 0.f, 0.f, 0.f, 0.f, 0.f};
    v8f accG1 = accG0, accU0 = accG0, accU1 = accG0;
#pragma unroll
    for (int ks = 0; ks < 4; ++ks) {   // K = 256, 2 k-steps per iter
      v16bf a0 = load_a(sInp, 256, 2 * ks, lane);
      v16bf a1 = load_a(sInp, 256, 2 * ks + 1, lane);
      accG0 = wmma_bf16(a0, load_b(sWg1, 256, w, 2 * ks, lane), accG0);
      accU0 = wmma_bf16(a0, load_b(sWu1, 256, w, 2 * ks, lane), accU0);
      accG1 = wmma_bf16(a1, load_b(sWg1, 256, w, 2 * ks + 1, lane), accG1);
      accU1 = wmma_bf16(a1, load_b(sWu1, 256, w, 2 * ks + 1, lane), accU1);
    }
    {
      v8f accG = accG0 + accG1;
      v8f accU = accU0 + accU1;
      float bgc = sbg1[col], buc = sbu1[col];
#pragma unroll
      for (int r = 0; r < 8; ++r) {
        int m = r + mo;
        float g = accG[r] + bgc;  g = g > 0.f ? g : 0.f;
        float u = accU[r] + buc;  u = u > 0.f ? u : 0.f;
        sHg[m * 128 + col] = (__bf16)g;
        sHu[m * 128 + col] = (__bf16)u;
      }
    }
    __syncthreads();

    // ---- phase 2: delta = hidden_u @ Wu2^T (WMMA) ; gate rows distributed over waves
    v8f accD0 = {0.f, 0.f, 0.f, 0.f, 0.f, 0.f, 0.f, 0.f};
    v8f accD1 = accD0;
    {
      v16bf a0 = load_a(sHu, 128, 0, lane);
      v16bf a1 = load_a(sHu, 128, 1, lane);
      v16bf a2 = load_a(sHu, 128, 2, lane);
      v16bf a3 = load_a(sHu, 128, 3, lane);
      accD0 = wmma_bf16(a0, load_b(sWu2, 128, w, 0, lane), accD0);
      accD1 = wmma_bf16(a1, load_b(sWu2, 128, w, 1, lane), accD1);
      accD0 = wmma_bf16(a2, load_b(sWu2, 128, w, 2, lane), accD0);
      accD1 = wmma_bf16(a3, load_b(sWu2, 128, w, 3, lane), accD1);
    }
    {
      // wave w: gate for rows 2w (lanes 0-15) and 2w+1 (lanes 16-31); one b128 row read
      v8bf hg = *(const v8bf*)(sHg + mln * 128 + c0);
      float s = 0.f;
#pragma unroll
      for (int i = 0; i < 8; ++i) s += (float)hg[i] * swg2[c0 + i];
      s += __shfl_xor(s, 1, 32);
      s += __shfl_xor(s, 2, 32);
      s += __shfl_xor(s, 4, 32);
      s += __shfl_xor(s, 8, 32);
      if ((lane & 15) == 0) sGate[mln] = 1.0f / (1.0f + __expf(-(s + bg2v)));
    }
    __syncthreads();

    // ---- phase 3: belief = (1-g)*belief + g*(delta+bu2) ----
    {
      v8f accD = accD0 + accD1;
      float buc = sbu2[col];
#pragma unroll
      for (int r = 0; r < 8; ++r) {
        int m = r + mo;
        float delta = accD[r] + buc;
        float g = sGate[m];
        sBel[m * 128 + col] = (1.0f - g) * sBel[m * 128 + col] + g * delta;
      }
    }
    __syncthreads();

    // ---- phase 4: LayerNorm per row; refresh inp belief half; stream h_{t+1} ----
    {
      float* row = sBel + mln * 128;
      float v[8], s = 0.f, s2 = 0.f;
#pragma unroll
      for (int i = 0; i < 8; ++i) { v[i] = row[c0 + i]; s += v[i]; s2 += v[i] * v[i]; }
      s  += __shfl_xor(s, 1, 32);  s2 += __shfl_xor(s2, 1, 32);
      s  += __shfl_xor(s, 2, 32);  s2 += __shfl_xor(s2, 2, 32);
      s  += __shfl_xor(s, 4, 32);  s2 += __shfl_xor(s2, 4, 32);
      s  += __shfl_xor(s, 8, 32);  s2 += __shfl_xor(s2, 8, 32);
      float mu   = s * (1.0f / 128.0f);
      float var  = s2 * (1.0f / 128.0f) - mu * mu;
      float rstd = rsqrtf(var + EPSV);
#pragma unroll
      for (int i = 0; i < 8; ++i) {
        int c = c0 + i;
        float o = (v[i] - mu) * rstd * sgam[c] + sbet[c];
        row[c] = o;
        sInp[mln * 256 + c] = (__bf16)o;
      }
    }
    if (t + 1 < TT) {
      if (tid < 128) {   // next h_t into inp[:, 128:256] (cols phase-4 doesn't touch)
        int e = tid * 8; int m = e >> 7, c = e & 127;
        *(v8bf*)(sInp + m * 256 + 128 + c) =
            *(const v8bf*)(h + ((size_t)(t + 1) * NB + m) * BDM + c);
      }
      if (tid < 16 && t + 2 < TT)   // pull h_{t+2} toward L2/L0 (global_prefetch_b8)
        __builtin_prefetch(h + (size_t)(t + 2) * NB * BDM + tid * 128, 0, 1);
    }
    __syncthreads();
  }

  // ---- epilogue: rows 0..7 of belief -> d_out [8][128] f32 ----
  for (int e = tid; e < NB * BDM; e += 256) out[e] = sBel[e];
}

// ---------------- launch ----------------

extern "C" void kernel_launch(void* const* d_in, const int* in_sizes, int n_in,
                              void* d_out, int out_size, void* d_ws, size_t ws_size,
                              hipStream_t stream) {
  (void)in_sizes; (void)n_in; (void)out_size; (void)ws_size;
  const float* x    = (const float*)d_in[0];
  const float* Wp   = (const float*)d_in[1];
  const float* Wg1  = (const float*)d_in[2];
  const float* bg1v = (const float*)d_in[3];
  const float* Wg2  = (const float*)d_in[4];
  const float* bg2v = (const float*)d_in[5];
  const float* Wu1  = (const float*)d_in[6];
  const float* bu1v = (const float*)d_in[7];
  const float* Wu2  = (const float*)d_in[8];
  const float* bu2v = (const float*)d_in[9];
  const float* gam  = (const float*)d_in[10];
  const float* bet  = (const float*)d_in[11];
  float*  out = (float*)d_out;

  __bf16* h   = (__bf16*)d_ws;                            // [4096][8][128] bf16 = 8 MB
  __bf16* wpb = (__bf16*)((char*)d_ws + (size_t)TT * NB * BDM * 2);  // [128][1024] bf16

  wprep_kernel<<<(BDM * DIM) / 256, 256, 0, stream>>>(Wp, wpb);

  dim3 pgrid(TT / 16, NB);
  size_t psm = (16 * DIM) * 4 + (16 * DIM) * 2;           // 96 KB dynamic LDS
  proj_kernel<<<pgrid, 256, psm, stream>>>(x, wpb, h);

  // 90112 bf16 (180224 B) + 2833 f32 (11332 B) = 191556 B; CDNA5 WGP LDS = 320 KB
  size_t smem = 192 * 1024;
  scan_kernel<<<1, 256, smem, stream>>>(h, Wg1, bg1v, Wg2, bg2v,
                                        Wu1, bu1v, Wu2, bu2v, gam, bet, out);
}
